// MultiHeadLatentAttention_63118839382030
// MI455X (gfx1250) — matrix operand order repaired and checked by reading, not compiled, not run on hardware
//
#include <hip/hip_runtime.h>
#include <cstdint>
#include <cstddef>

// ---------------------------------------------------------------------------
// MI455X (gfx1250) MLA forward: bf16 WMMA GEMMs + flash attention.
// wave32; v_wmma_f32_16x16x32_bf16 everywhere; WGP-scope global_prefetch_b8
// via inline asm (working set is GL2-resident; only the GL2->WGP hop needs
// prefetching). Hot loops force-unrolled x2 with compile-time trip counts so
// the unroller cannot bail on the volatile asm.
// ---------------------------------------------------------------------------

typedef __attribute__((ext_vector_type(16))) __bf16 v16bf;
typedef __attribute__((ext_vector_type(8)))  float  v8f;

union Frag16 {               // one WMMA bf16 operand: 16 bf16 = 32B = 8 VGPRs
  v16bf v;
  uint4 q[2];
  unsigned short us[16];
};

#define DEVFN static __device__ __forceinline__

DEVFN unsigned short f32_to_bf16(float f) {      // round-to-nearest-even
  unsigned int u = __float_as_uint(f);
  u += 0x7fffu + ((u >> 16) & 1u);
  return (unsigned short)(u >> 16);
}

DEVFN v8f wmma_bf16(const Frag16& a, const Frag16& b, v8f c) {
  return __builtin_amdgcn_wmma_f32_16x16x32_bf16(
      /*neg_a=*/false, a.v, /*neg_b=*/false, b.v,
      /*c_mod=*/(short)0, c, /*reuse_a=*/false, /*reuse_b=*/false);
}

DEVFN v8f v8f_zero() { return (v8f){0.f,0.f,0.f,0.f,0.f,0.f,0.f,0.f}; }

// WGP-scope prefetch: pulls line into all cache levels (ISA 10.5, scope 0).
// No "memory" clobber: a prefetch has no architecturally visible effect.
DEVFN void prefetch_wgp(const void* p) {
  unsigned long long a = (unsigned long long)p;
  asm volatile("global_prefetch_b8 %0, off" :: "v"(a));
}

// ---------------------------------------------------------------------------
// Vectorized f32 -> bf16 conversion (4 elements / thread)
// ---------------------------------------------------------------------------
__global__ void cvt_f32_bf16_v4(const float4* __restrict__ in,
                                uint2* __restrict__ out, int n4) {
  int i = blockIdx.x * blockDim.x + threadIdx.x;
  if (i < n4) {
    float4 f = in[i];
    uint2 o;
    o.x = (unsigned)f32_to_bf16(f.x) | ((unsigned)f32_to_bf16(f.y) << 16);
    o.y = (unsigned)f32_to_bf16(f.z) | ((unsigned)f32_to_bf16(f.w) << 16);
    out[i] = o;
  }
}

// ---------------------------------------------------------------------------
// GEMM: C[M][N] = A[M][K] * B[K][N], A/B bf16 row-major, K compile-time.
// Block = 256 threads = 8 waves arranged 4(M) x 2(N); block tile 128x128.
// Each wave owns a 32x64 tile: 2 A-frags, 8 accumulators, 8 WMMA / k-step.
// Grid: x = N/128, y = M/128.
// OUT_MODE: 0 = bf16 row-major, 1 = bf16 transposed (C^T, ldc = M), 2 = f32.
// ---------------------------------------------------------------------------
template <int OUT_MODE, int K>
__global__ __launch_bounds__(256) void gemm_bf16_wmma(
    const unsigned short* __restrict__ A, int lda,
    const unsigned short* __restrict__ B, int ldb,
    void* __restrict__ Cv, int ldc) {
  const int lane = threadIdx.x & 31;
  const int wave = threadIdx.x >> 5;
  const int half = lane >> 4;
  const int l16  = lane & 15;
  const int wm   = wave >> 1;                       // 0..3
  const int wn   = wave & 1;                        // 0..1
  const int mtile = blockIdx.y * 128 + wm * 32;     // 32 rows per wave
  const int nbase = blockIdx.x * 128 + wn * 64;     // 64 cols per wave

  v8f acc[2][4];
#pragma unroll
  for (int s = 0; s < 2; ++s)
#pragma unroll
    for (int j = 0; j < 4; ++j) acc[s][j] = v8f_zero();

  const unsigned short* arow0 = A + (size_t)(mtile + l16) * lda + half * 8;
  const unsigned short* arow1 = arow0 + (size_t)16 * lda;

#pragma unroll 2
  for (int kb = 0; kb < K; kb += 32) {
    Frag16 a0, a1;
    a0.q[0] = *(const uint4*)(arow0 + kb);
    a0.q[1] = *(const uint4*)(arow0 + kb + 16);
    a1.q[0] = *(const uint4*)(arow1 + kb);
    a1.q[1] = *(const uint4*)(arow1 + kb + 16);

    const unsigned short* brow = B + (size_t)(kb + lane) * ldb + nbase;
    // WGP-scope prefetch of the next k-block (data is GL2-resident; we only
    // need to hide the GL2->WGP hop)
    prefetch_wgp(arow0 + kb + 32);
    prefetch_wgp(brow + (size_t)32 * ldb);

#pragma unroll
    for (int j = 0; j < 4; ++j) {                   // B-frag: lane=n, 16 k (table order)
      Frag16 b;
#pragma unroll
      for (int s = 0; s < 16; ++s) {
        const int kk = kb + (s < 8 ? half * 8 + s : 16 + half * 8 + (s - 8));
        b.us[s] = B[(size_t)kk * ldb + nbase + j * 16 + l16];
      }
      acc[0][j] = wmma_bf16(a0, b, acc[0][j]);
      acc[1][j] = wmma_bf16(a1, b, acc[1][j]);
    }
  }

#pragma unroll
  for (int s = 0; s < 2; ++s) {
    const int mb = mtile + s * 16 + half * 8;
#pragma unroll
    for (int j = 0; j < 4; ++j) {
      if (OUT_MODE == 2) {
        float* C = (float*)Cv;
#pragma unroll
        for (int r = 0; r < 8; ++r)
          C[(size_t)(mb + r) * ldc + nbase + j * 16 + l16] = acc[s][j][r];
      } else if (OUT_MODE == 0) {
        unsigned short* C = (unsigned short*)Cv;
#pragma unroll
        for (int r = 0; r < 8; ++r)
          C[(size_t)(mb + r) * ldc + nbase + j * 16 + l16] =
              f32_to_bf16(acc[s][j][r]);
      } else {  // transposed: lane's 8 rows contiguous -> one b128 store
        unsigned short* C = (unsigned short*)Cv;
        union { unsigned short us[8]; uint4 q; } t;
#pragma unroll
        for (int r = 0; r < 8; ++r) t.us[r] = f32_to_bf16(acc[s][j][r]);
        *(uint4*)&C[(size_t)(nbase + j * 16 + l16) * ldc + mb] = t.q;
      }
    }
  }
}

// ---------------------------------------------------------------------------
// Flash attention (GQA, non-causal, key padding mask).
//   Qb  : [4096][2048] bf16  (token-major, head col = head*128)
//   Ktb : [512][4096]  bf16  (K transposed: row = hkv*128+d, col = b*2048+s)
//   Vb  : [4096][512]  bf16  (token-major, col = hkv*128+d)
//   mask: [2][2048] int
//   Ob  : [4096][2048] bf16
// One wave per (b, head, 16-query tile): 2*16*128 = 4096 waves.
// ---------------------------------------------------------------------------
#define S_LEN 2048
#define DHEAD 128
#define ATT_SCALE 0.088388347648318447f  // 1/sqrt(128)

__global__ __launch_bounds__(128) void mla_flash_attn(
    const unsigned short* __restrict__ Qb,
    const unsigned short* __restrict__ Ktb,
    const unsigned short* __restrict__ Vb,
    const int* __restrict__ mask,
    unsigned short* __restrict__ Ob) {
  __shared__ unsigned short plds[4][16][32] __attribute__((aligned(16)));

  const int lane = threadIdx.x & 31;
  const int wave = threadIdx.x >> 5;
  const int half = lane >> 4;
  const int l16  = lane & 15;

  const int task = blockIdx.x * 4 + wave;
  const int qt   = task & 127;
  const int head = (task >> 7) & 15;
  const int b    = task >> 11;
  const int hkv  = head >> 2;
  const int tokbase = b * S_LEN + qt * 16;

  // Resident Q fragments: 16 rows x 128 dh, 4 K-chunks of 32
  Frag16 aq[4];
  const unsigned short* qrow =
      Qb + (size_t)(tokbase + l16) * 2048 + head * DHEAD + half * 8;
#pragma unroll
  for (int c = 0; c < 4; ++c) {
    aq[c].q[0] = *(const uint4*)(qrow + c * 32);
    aq[c].q[1] = *(const uint4*)(qrow + c * 32 + 16);
  }

  v8f o[8];
#pragma unroll
  for (int t = 0; t < 8; ++t) o[t] = v8f_zero();
  float mrun[8], lrun[8];
#pragma unroll
  for (int r = 0; r < 8; ++r) { mrun[r] = -1e30f; lrun[r] = 0.f; }

#pragma unroll 2
  for (int kb = 0; kb < S_LEN; kb += 32) {
    // ---- scores: two 16x16 tiles over 32 keys, contract over dh=128 ----
    v8f s0 = v8f_zero();
    v8f s1 = v8f_zero();
#pragma unroll
    for (int dc = 0; dc < 4; ++dc) {
      const unsigned short* kr =
          Ktb + (size_t)(hkv * DHEAD + dc * 32) * 4096 + b * S_LEN + kb + l16;
      Frag16 bk0, bk1;
#pragma unroll
      for (int s = 0; s < 16; ++s) {
        const int kk = (s < 8 ? half * 8 + s : 16 + half * 8 + (s - 8));
        bk0.us[s] = kr[(size_t)kk * 4096];
        bk1.us[s] = kr[(size_t)kk * 4096 + 16];
      }
      s0 = wmma_bf16(aq[dc], bk0, s0);
      s1 = wmma_bf16(aq[dc], bk1, s1);
    }

    const int mk0 = mask[b * S_LEN + kb + l16];
    const int mk1 = mask[b * S_LEN + kb + 16 + l16];

    // WGP-scope prefetch of next key-block of V (new cache lines; the Kt
    // stream advances within a row and rides the same lines)
    const unsigned short* vrow =
        Vb + (size_t)(b * S_LEN + kb + lane) * 512 + hkv * DHEAD;
    prefetch_wgp(vrow + (size_t)32 * 512);

    // ---- online softmax (rows live per half-wave: half0=r, half1=8+r) ----
    float p0[8], p1[8], alpha[8];
#pragma unroll
    for (int r = 0; r < 8; ++r) {
      float a0 = s0[r] * ATT_SCALE; if (mk0 == 0) a0 = -1e30f;
      float a1 = s1[r] * ATT_SCALE; if (mk1 == 0) a1 = -1e30f;
      float mx = fmaxf(a0, a1);
#pragma unroll
      for (int off = 8; off >= 1; off >>= 1)
        mx = fmaxf(mx, __shfl_xor(mx, off, 16));
      const float mnew = fmaxf(mrun[r], mx);
      alpha[r] = __expf(mrun[r] - mnew);
      const float e0 = __expf(a0 - mnew);
      const float e1 = __expf(a1 - mnew);
      float sum = e0 + e1;
#pragma unroll
      for (int off = 8; off >= 1; off >>= 1) sum += __shfl_xor(sum, off, 16);
      lrun[r] = lrun[r] * alpha[r] + sum;
      mrun[r] = mnew;
      p0[r] = e0; p1[r] = e1;
    }
#pragma unroll
    for (int t = 0; t < 8; ++t)
#pragma unroll
      for (int r = 0; r < 8; ++r) o[t][r] *= alpha[r];

    // ---- C-layout P -> A-layout via per-wave LDS (wave-private slice) ----
#pragma unroll
    for (int r = 0; r < 8; ++r) {
      const int row = half * 8 + r;
      plds[wave][row][l16]      = f32_to_bf16(p0[r]);
      plds[wave][row][l16 + 16] = f32_to_bf16(p1[r]);
    }
    Frag16 apf;
    apf.q[0] = *(const uint4*)&plds[wave][l16][half * 8];
    apf.q[1] = *(const uint4*)&plds[wave][l16][16 + half * 8];

    // ---- O += P(16x32) * V(32x128): V gathered per lane column, k table order ----
#pragma unroll
    for (int t = 0; t < 8; ++t) {
      Frag16 bv;
#pragma unroll
      for (int s = 0; s < 16; ++s) {
        const int kk = (s < 8 ? half * 8 + s : 16 + half * 8 + (s - 8));
        bv.us[s] = Vb[(size_t)(b * S_LEN + kb + kk) * 512 + hkv * DHEAD + t * 16 + l16];
      }
      o[t] = wmma_bf16(apf, bv, o[t]);
    }
  }

  // ---- epilogue: normalize and store bf16 ----
#pragma unroll
  for (int t = 0; t < 8; ++t)
#pragma unroll
    for (int r = 0; r < 8; ++r) {
      const float val = o[t][r] / lrun[r];
      Ob[(size_t)(tokbase + half * 8 + r) * 2048 + head * DHEAD + t * 16 + l16] =
          f32_to_bf16(val);
    }
}

// ---------------------------------------------------------------------------
// Host launcher
// ---------------------------------------------------------------------------
extern "C" void kernel_launch(void* const* d_in, const int* in_sizes, int n_in,
                              void* d_out, int out_size, void* d_ws,
                              size_t ws_size, hipStream_t stream) {
  (void)in_sizes; (void)n_in; (void)out_size; (void)ws_size;

  const float* Q_input = (const float*)d_in[0];  // [2,2048,2048]
  const float* Wq      = (const float*)d_in[1];  // [2048,2048]
  const float* Wkvb    = (const float*)d_in[2];  // [2048,1024]
  const float* Wkva    = (const float*)d_in[3];  // [512,1024]
  const float* Wo      = (const float*)d_in[4];  // [2048,2048]
  const int*   amask   = (const int*)d_in[5];    // [2,2048]
  float* out = (float*)d_out;                    // [2,2048,2048]

  char* ws = (char*)d_ws;
  size_t off = 0;
  auto carve = [&](size_t bytes) {
    void* p = ws + off;
    off += (bytes + 255) & ~(size_t)255;
    return p;
  };
  unsigned short* Qin_b  = (unsigned short*)carve(4096ull * 2048 * 2);
  unsigned short* Wq_b   = (unsigned short*)carve(2048ull * 2048 * 2);
  unsigned short* Wkvb_b = (unsigned short*)carve(2048ull * 1024 * 2);
  unsigned short* Wkva_b = (unsigned short*)carve(512ull  * 1024 * 2);
  unsigned short* Wo_b   = (unsigned short*)carve(2048ull * 2048 * 2);
  unsigned short* Qb     = (unsigned short*)carve(4096ull * 2048 * 2);
  unsigned short* kvb_b  = (unsigned short*)carve(4096ull * 1024 * 2);
  unsigned short* Kt_b   = (unsigned short*)carve(512ull  * 4096 * 2);
  unsigned short* V_b    = (unsigned short*)carve(4096ull * 512  * 2);
  unsigned short* attn_b = (unsigned short*)carve(4096ull * 2048 * 2);

  auto cvt = [&](const float* in, unsigned short* o, long long n) {
    long long n4 = n / 4;
    cvt_f32_bf16_v4<<<dim3((unsigned)((n4 + 255) / 256)), dim3(256), 0,
                      stream>>>((const float4*)in, (uint2*)o, (int)n4);
  };
  cvt(Q_input, Qin_b, 4096ll * 2048);
  cvt(Wq,      Wq_b,   2048ll * 2048);
  cvt(Wkvb,    Wkvb_b, 2048ll * 1024);
  cvt(Wkva,    Wkva_b, 512ll  * 1024);
  cvt(Wo,      Wo_b,   2048ll * 2048);

  // Q = Qin @ Wq                     (4096x2048x2048)
  gemm_bf16_wmma<0, 2048><<<dim3(2048 / 128, 4096 / 128), 256, 0, stream>>>(
      Qin_b, 2048, Wq_b, 2048, Qb, 2048);
  // kvb = Qin @ Wkvb                 (4096x1024x2048)
  gemm_bf16_wmma<0, 2048><<<dim3(1024 / 128, 4096 / 128), 256, 0, stream>>>(
      Qin_b, 2048, Wkvb_b, 1024, kvb_b, 1024);
  // Kt = (k_b @ Wkva[:, :512])^T     (transposed epilogue, ldc = 4096)
  gemm_bf16_wmma<1, 512><<<dim3(512 / 128, 4096 / 128), 256, 0, stream>>>(
      kvb_b, 1024, Wkva_b, 1024, Kt_b, 4096);
  // V  = v_b @ Wkva[:, 512:]         (row-major)
  gemm_bf16_wmma<0, 512><<<dim3(512 / 128, 4096 / 128), 256, 0, stream>>>(
      kvb_b + 512, 1024, Wkva_b + 512, 1024, V_b, 512);
  // flash attention: 4096 wave-tasks, 4 waves/block
  mla_flash_attn<<<dim3(1024), dim3(128), 0, stream>>>(Qb, Kt_b, V_b, amask,
                                                       attn_b);
  // out = attn @ Wo (f32 epilogue)   (4096x2048x2048)
  gemm_bf16_wmma<2, 2048><<<dim3(2048 / 128, 4096 / 128), 256, 0, stream>>>(
      attn_b, 2048, Wo_b, 2048, out, 2048);
}
